// VectorQuantizer_317827580710
// MI455X (gfx1250) — matrix operand order repaired and checked by reading, not compiled, not run on hardware
//
#include <hip/hip_runtime.h>
#include <math.h>

typedef __attribute__((ext_vector_type(16))) _Float16 v16h;
typedef __attribute__((ext_vector_type(8)))  _Float16 v8h;
typedef __attribute__((ext_vector_type(8)))  float    v8f;

#define K_CODES   1024
#define D_DIM     256
#define N_ROWS    32768
#define LDS_STRIDE 264            // 256 + 8 halfs pad -> 528B row stride, 16B aligned
#define B_STRIDE   264            // same padding for the B tile
#define LDS_A_BYTES (128 * LDS_STRIDE * 2)       // 67584
#define LDS_B_BYTES (32 * B_STRIDE * 2)          // 16896 per buffer

// workspace layout (bytes)
#define WS_WH    0                // _Float16 [1024*256] = 512KB
#define WS_WSUM  (512*1024)       // float [1024]
#define WS_IDX   (WS_WSUM + 4096)    // int [32768]
#define WS_CNT   (WS_IDX + 131072)   // int [1024]
#define WS_LOSS  (WS_CNT + 4096)     // float [1]

// ---------------------------------------------------------------------------
// Kernel 0: codebook prep: f32 -> f16 copy, ||e||^2 per code, zero counters.
// grid = 1024 (one block per code), block = 256 (one thread per dim)
// ---------------------------------------------------------------------------
__global__ void vq_prep(const float* __restrict__ W, _Float16* __restrict__ Wh,
                        float* __restrict__ wsum, int* __restrict__ counts,
                        float* __restrict__ lossacc) {
    const int k = blockIdx.x, t = threadIdx.x;
    const float v = W[k * D_DIM + t];
    Wh[k * D_DIM + t] = (_Float16)v;
    float s = v * v;
    #pragma unroll
    for (int m = 16; m >= 1; m >>= 1) s += __shfl_xor(s, m, 32);
    __shared__ float red[8];
    const int wid = t >> 5, lane = t & 31;
    if (lane == 0) red[wid] = s;
    __syncthreads();
    if (t == 0) {
        float tot = 0.f;
        #pragma unroll
        for (int i = 0; i < 8; ++i) tot += red[i];
        wsum[k]   = tot;
        counts[k] = 0;
        if (k == 0) lossacc[0] = 0.f;
    }
}

// ---------------------------------------------------------------------------
// Kernel 1: distances via f16 WMMA + per-row argmin over all 1024 codes.
// grid = 256 blocks (128 rows each), block = 256 threads = 8 waves.
// Dynamic LDS: A tile 128x264 halfs + ping/pong B tiles Bb0/Bb1 (static
// pointers only -> everything stays addrspace(3) = ds_load/ds_store).
// Per phase: issue global_load_b128 for the NEXT code tile, run the WMMA
// chain off the CURRENT LDS buffer, then drain the loads into the other
// buffer and barrier. argmin(dist) = argmin(||e||^2 - 2 x.e).
// ---------------------------------------------------------------------------
__global__ void vq_argmin(const float* __restrict__ X, const _Float16* __restrict__ Wh,
                          const float* __restrict__ wsum, int* __restrict__ idx,
                          int* __restrict__ counts) {
    extern __shared__ char smem_raw[];
    _Float16* Ah  = (_Float16*)smem_raw;
    _Float16* Bb0 = (_Float16*)(smem_raw + LDS_A_BYTES);
    _Float16* Bb1 = (_Float16*)(smem_raw + LDS_A_BYTES + LDS_B_BYTES);

    const int t    = threadIdx.x;
    const int tile = blockIdx.x;          // 256 tiles of 128 rows; 1024 % 128 == 0
    const int b    = tile >> 3;           // batch index
    const int hw0  = (tile & 7) * 128;    // spatial offset within batch image
    const float* Xb = X + (size_t)b * (D_DIM * 1024) + hw0;

    // Stage A tile: 128 rows (n) x 256 dims, converting f32 -> f16.
    #pragma unroll 4
    for (int i = 0; i < 128; ++i) {
        const int e = t + 256 * i;
        const int d = e >> 7;             // 0..255
        const int n = e & 127;            // 0..127
        Ah[n * LDS_STRIDE + d] = (_Float16)Xb[d * 1024 + n];
    }

    // B staging: this thread's 4 fixed 16-byte segments of a 32x256 f16 tile.
    const int sc0 = t >> 5;               // codes handled: sc0, sc0+8, +16, +24
    const int sseg = t & 31;              // 8-half segment within the row
    v8h breg[4];
    auto stage_load = [&](int k0n) {
        const _Float16* src = Wh + (size_t)(k0n + sc0) * D_DIM + sseg * 8;
        #pragma unroll
        for (int i = 0; i < 4; ++i)
            breg[i] = *(const v8h*)(src + i * 8 * D_DIM);   // rows sc0 + 8*i
    };
    auto stage_store = [&](_Float16* dst) {
        #pragma unroll
        for (int i = 0; i < 4; ++i)
            *(v8h*)(dst + (sc0 + 8 * i) * B_STRIDE + sseg * 8) = breg[i];
    };

    stage_load(0);
    stage_store(Bb0);
    __syncthreads();

    const int wid  = t >> 5;              // wave 0..7 -> 16-row strip
    const int lane = t & 31;
    const int lm   = lane & 15;
    const int hi   = lane >> 4;
    // A fragment (contiguous k order, matching B): lanes 0-15 hold K{0..15},
    // lanes 16-31 hold K{16..31} of row M = lane%16.
    const _Float16* arow = Ah + (16 * wid + lm) * LDS_STRIDE + 16 * hi;

    float minv[8]; int mini[8];
    #pragma unroll
    for (int j = 0; j < 8; ++j) { minv[j] = 3.4e38f; mini[j] = 0; }

    auto compute = [&](const _Float16* bbase, int k0) {
        v8f acc0 = {}; v8f acc1 = {};
        const _Float16* b0p = bbase + lm * B_STRIDE + 16 * hi;
        const _Float16* b1p = b0p + 16 * B_STRIDE;
        #pragma unroll
        for (int d0 = 0; d0 < D_DIM; d0 += 32) {
            const v8h alo = *(const v8h*)(arow + d0);
            const v8h ahi = *(const v8h*)(arow + d0 + 8);
            const v16h a  = __builtin_shufflevector(alo, ahi,
                                0,1,2,3,4,5,6,7,8,9,10,11,12,13,14,15);
            const v16h b0 = *(const v16h*)(b0p + d0);
            const v16h b1 = *(const v16h*)(b1p + d0);
            acc0 = __builtin_amdgcn_wmma_f32_16x16x32_f16(false, a, false, b0,
                                                          (short)0, acc0, false, false);
            acc1 = __builtin_amdgcn_wmma_f32_16x16x32_f16(false, a, false, b1,
                                                          (short)0, acc1, false, false);
        }
        const float ws0 = wsum[k0 + lm];
        const float ws1 = wsum[k0 + 16 + lm];
        // C layout: VGPR j = row j (lanes 0-15) / row j+8 (lanes 16-31)
        #pragma unroll
        for (int j = 0; j < 8; ++j) {
            const float dv0 = ws0 - 2.f * acc0[j];
            if (dv0 < minv[j]) { minv[j] = dv0; mini[j] = k0 + lm; }
            const float dv1 = ws1 - 2.f * acc1[j];
            if (dv1 < minv[j]) { minv[j] = dv1; mini[j] = k0 + 16 + lm; }
        }
    };

    for (int k0 = 0; k0 < K_CODES; k0 += 64) {
        // Phase A: compute from Bb0, stream tile (k0+32) into Bb1.
        stage_load(k0 + 32);                 // k0+32 <= 992 < 1024 always
        compute(Bb0, k0);
        stage_store(Bb1);
        __syncthreads();
        // Phase B: compute from Bb1, stream tile (k0+64) into Bb0.
        const bool more = (k0 + 64) < K_CODES;
        if (more) stage_load(k0 + 64);
        compute(Bb1, k0 + 32);
        if (more) stage_store(Bb0);
        __syncthreads();
    }

    // Reduce min across the 16 column-lanes of each half-wave.
    #pragma unroll
    for (int m = 1; m <= 8; m <<= 1) {
        #pragma unroll
        for (int j = 0; j < 8; ++j) {
            const float ov = __shfl_xor(minv[j], m, 32);
            const int   oi = __shfl_xor(mini[j], m, 32);
            if (ov < minv[j]) { minv[j] = ov; mini[j] = oi; }
        }
    }
    if (lm == 0) {
        const int nbase = tile * 128 + wid * 16 + hi * 8;
        #pragma unroll
        for (int j = 0; j < 8; ++j) {
            idx[nbase + j] = mini[j];
            atomicAdd(&counts[mini[j]], 1);
        }
    }
}

// ---------------------------------------------------------------------------
// Kernel 2: gather codes (fp32), write quantized_st [B,C,H,W], sum (q-x)^2.
// grid = 512 (32 b x 16 hw-chunks of 64), block = 256.
// ---------------------------------------------------------------------------
__global__ void vq_quantize(const float* __restrict__ X, const float* __restrict__ W,
                            const int* __restrict__ idx, float* __restrict__ outq,
                            float* __restrict__ lossacc) {
    const int t   = threadIdx.x;
    const int b   = blockIdx.x >> 4;
    const int hw0 = (blockIdx.x & 15) * 64;
    __shared__ int lidx[64];
    if (t < 64) lidx[t] = idx[b * 1024 + hw0 + t];
    __syncthreads();

    const int hwo  = t & 63;
    const int csub = t >> 6;                 // 0..3
    const int k    = lidx[hwo];
    const float* xb = X    + (size_t)b * (D_DIM * 1024) + hw0 + hwo;
    float*       ob = outq + (size_t)b * (D_DIM * 1024) + hw0 + hwo;
    const float* wr = W + k * D_DIM;

    float s = 0.f;
    #pragma unroll 4
    for (int it = 0; it < 64; ++it) {
        const int c = it * 4 + csub;
        const float q = wr[c];
        const float x = xb[c * 1024];
        const float d = q - x;
        s += d * d;
        ob[c * 1024] = q;                     // straight-through value == quantized
    }
    #pragma unroll
    for (int m = 16; m >= 1; m >>= 1) s += __shfl_xor(s, m, 32);
    __shared__ float red[8];
    const int wid = t >> 5, lane = t & 31;
    if (lane == 0) red[wid] = s;
    __syncthreads();
    if (t == 0) {
        float tot = 0.f;
        #pragma unroll
        for (int i = 0; i < 8; ++i) tot += red[i];
        atomicAdd(lossacc, tot);
    }
}

// ---------------------------------------------------------------------------
// Kernel 3: one-hot encodings [32768, 1024]. One block per row; each thread
// writes 4 floats as two float2 (base is only 8B-aligned).
// ---------------------------------------------------------------------------
__global__ void vq_encodings(const int* __restrict__ idx, float* __restrict__ enc) {
    const int row = blockIdx.x;
    const int t   = threadIdx.x;
    const int k   = idx[row];
    float2 a = make_float2(0.f, 0.f), c2 = make_float2(0.f, 0.f);
    if ((k >> 2) == t) {
        const int c = k & 3;
        if      (c == 0) a.x  = 1.f;
        else if (c == 1) a.y  = 1.f;
        else if (c == 2) c2.x = 1.f;
        else             c2.y = 1.f;
    }
    float2* p = (float2*)(enc + (size_t)row * 1024 + t * 4);
    p[0] = a; p[1] = c2;
}

// ---------------------------------------------------------------------------
// Kernel 4: loss + perplexity scalars. Single block of 256 threads.
// ---------------------------------------------------------------------------
__global__ void vq_finalize(const int* __restrict__ counts, const float* __restrict__ lossacc,
                            float* __restrict__ out_loss, float* __restrict__ out_perp) {
    const int t = threadIdx.x;
    float s = 0.f;
    #pragma unroll
    for (int i = 0; i < 4; ++i) {
        const int k = t + 256 * i;
        const float p = (float)counts[k] * (1.f / 32768.f);
        s -= p * logf(p + 1e-10f);
    }
    #pragma unroll
    for (int m = 16; m >= 1; m >>= 1) s += __shfl_xor(s, m, 32);
    __shared__ float red[8];
    const int wid = t >> 5, lane = t & 31;
    if (lane == 0) red[wid] = s;
    __syncthreads();
    if (t == 0) {
        float tot = 0.f;
        #pragma unroll
        for (int i = 0; i < 8; ++i) tot += red[i];
        out_perp[0] = expf(tot);
        out_loss[0] = 1.25f * lossacc[0] * (1.f / 8388608.f);  // (1+0.25)*MSE
    }
}

// ---------------------------------------------------------------------------
// d_out layout (float): [0]=loss, [1..8388609)=quantized, [8388609]=perplexity,
// [8388610..)=encodings (32768*1024).
// ---------------------------------------------------------------------------
extern "C" void kernel_launch(void* const* d_in, const int* in_sizes, int n_in,
                              void* d_out, int out_size, void* d_ws, size_t ws_size,
                              hipStream_t stream) {
    const float* X = (const float*)d_in[0];   // [32, 256, 32, 32]
    const float* W = (const float*)d_in[1];   // [1024, 256]
    float* out = (float*)d_out;
    char*  ws  = (char*)d_ws;

    _Float16* Wh     = (_Float16*)(ws + WS_WH);
    float*    wsum   = (float*)(ws + WS_WSUM);
    int*      idx    = (int*)(ws + WS_IDX);
    int*      counts = (int*)(ws + WS_CNT);
    float*    lossac = (float*)(ws + WS_LOSS);

    vq_prep<<<dim3(K_CODES), dim3(256), 0, stream>>>(W, Wh, wsum, counts, lossac);

    const size_t lds_bytes = LDS_A_BYTES + 2 * LDS_B_BYTES;   // 101376
    vq_argmin<<<dim3(N_ROWS / 128), dim3(256), lds_bytes, stream>>>(X, Wh, wsum, idx, counts);

    vq_quantize<<<dim3(512), dim3(256), 0, stream>>>(X, W, idx, out + 1, lossac);

    vq_encodings<<<dim3(N_ROWS), dim3(256), 0, stream>>>(idx, out + 8388610);

    vq_finalize<<<dim3(1), dim3(256), 0, stream>>>(counts, lossac, out, out + 8388609);
}
